// StochasticLSTMCell_27754078667124
// MI455X (gfx1250) — compile-verified, matrix operand order checked by reading, not run on hardware
//
#include <hip/hip_runtime.h>
#include <hip/hip_bf16.h>

// ---------------------------------------------------------------------------
// Types
// ---------------------------------------------------------------------------
typedef __attribute__((ext_vector_type(8)))  __bf16       v8bf;
typedef __attribute__((ext_vector_type(16))) __bf16       v16bf;
typedef __attribute__((ext_vector_type(8)))  float        v8f;
typedef __attribute__((ext_vector_type(4)))  unsigned int u32x4;
typedef __attribute__((ext_vector_type(8)))  int          i32x8;
typedef __attribute__((ext_vector_type(4)))  int          i32x4;

#define T_STEPS 512
#define BATCH   64
#define IDIM    256
#define HDIM    512
#define NCOLS   2048   // 4*HDIM
#define NBLK    16     // persistent blocks in recurrent kernel

// ---------------------------------------------------------------------------
// Fragment loader: 16-bit A/B matrix fragment (16 rows x 32 K) from a
// row-major [rows][ld] bf16 array (global or LDS). ISA 7.12.2 layout:
//   lanes 0-15 : row = row0+lane,    K = {0..7, 16..23}
//   lanes 16-31: row = row0+lane-16, K = {8..15, 24..31}
// ---------------------------------------------------------------------------
__device__ __forceinline__ v16bf load_frag_rm(const __bf16* base,
                                              int row0, int ld, int k0, int lane) {
  const int half = (lane >> 4) & 1;
  const int r    = row0 + (lane & 15);
  const __bf16* p = base + (size_t)r * (size_t)ld + (size_t)(k0 + half * 8);
  v8bf lo = *(const v8bf*)(p);
  v8bf hi = *(const v8bf*)(p + 16);
  return __builtin_shufflevector(lo, hi, 0,1,2,3,4,5,6,7,8,9,10,11,12,13,14,15);
}

__device__ __forceinline__ float fast_sigmoid(float x) {
  return 1.0f / (1.0f + __expf(-x));
}
__device__ __forceinline__ float fast_tanh(float x) {
  return 1.0f - 2.0f / (__expf(2.0f * x) + 1.0f);
}

// ---------------------------------------------------------------------------
// Kernel 0: zero the sync counter and hm ping buffer (read at t=0).
// ---------------------------------------------------------------------------
__global__ void init_kernel(__bf16* hmbuf, unsigned* syncc) {
  const int i = blockIdx.x * 256 + threadIdx.x;
  if (i == 0) *syncc = 0u;
  if (i < BATCH * HDIM) hmbuf[i] = (__bf16)0.0f;
}

// ---------------------------------------------------------------------------
// Kernel 1: bf16 operand prep (xb = x*zx, Wxb/Whb copies, bsum = bx+bh).
// ---------------------------------------------------------------------------
__global__ void convert_kernel(const float* __restrict__ x,
                               const float* __restrict__ zx,
                               const float* __restrict__ Wx,
                               const float* __restrict__ bx,
                               const float* __restrict__ Wh,
                               const float* __restrict__ bh,
                               __bf16* __restrict__ xb,
                               __bf16* __restrict__ Wxb,
                               __bf16* __restrict__ Whb,
                               float*  __restrict__ bsum) {
  const size_t NXB  = (size_t)T_STEPS * BATCH * IDIM;
  const size_t NWX  = (size_t)4 * HDIM * IDIM;
  const size_t NWH  = (size_t)4 * HDIM * HDIM;
  const size_t TOT  = NXB + NWX + NWH + NCOLS;
  for (size_t idx = (size_t)blockIdx.x * blockDim.x + threadIdx.x;
       idx < TOT; idx += (size_t)gridDim.x * blockDim.x) {
    if (idx < NXB) {
      xb[idx] = (__bf16)(x[idx] * zx[idx % IDIM]);
    } else if (idx < NXB + NWX) {
      const size_t j = idx - NXB;
      Wxb[j] = (__bf16)Wx[j];
    } else if (idx < NXB + NWX + NWH) {
      const size_t j = idx - NXB - NWX;
      Whb[j] = (__bf16)Wh[j];
    } else {
      const size_t n = idx - NXB - NWX - NWH;
      bsum[n] = bx[n] + bh[n];
    }
  }
}

// ---------------------------------------------------------------------------
// Kernel 2: xp[m][n] = sum_k xb[m][k]*Wxb[n][k]  (M=32768, N=2048, K=256)
// Register-blocked: each wave computes 4(M) x 2(N) tiles -> per K-chunk:
// 4 A-frag + 2 B-frag loads feed 8 WMMAs (cuts L2 re-reads ~3x vs 1 tile).
// ---------------------------------------------------------------------------
__global__ __launch_bounds__(256) void gemm_xp_kernel(const __bf16* __restrict__ xb,
                                                      const __bf16* __restrict__ Wxb,
                                                      float* __restrict__ xp) {
  const int lane = threadIdx.x & 31;
  const int wave = threadIdx.x >> 5;
  const int unit = blockIdx.x * 8 + wave;   // 32768 units (4x2 tiles each)
  const int uN   = unit & 63;               // 64 units across N (2 tiles each)
  const int uM   = unit >> 6;               // 512 units across M (4 tiles each)
  const int mBase = uM * 64;
  const int nBase = uN * 32;

  v8f acc[4][2];
#pragma unroll
  for (int i = 0; i < 4; ++i)
#pragma unroll
    for (int n = 0; n < 2; ++n)
      acc[i][n] = (v8f){0.f,0.f,0.f,0.f,0.f,0.f,0.f,0.f};

#pragma unroll
  for (int kk = 0; kk < IDIM; kk += 32) {
    v16bf a[4];
#pragma unroll
    for (int i = 0; i < 4; ++i)
      a[i] = load_frag_rm(xb, mBase + i * 16, IDIM, kk, lane);
#pragma unroll
    for (int n = 0; n < 2; ++n) {
      v16bf b = load_frag_rm(Wxb, nBase + n * 16, IDIM, kk, lane);
#pragma unroll
      for (int i = 0; i < 4; ++i)
        acc[i][n] = __builtin_amdgcn_wmma_f32_16x16x32_bf16(false, a[i], false, b,
                                                            (short)0, acc[i][n],
                                                            false, false);
    }
  }

  const int mOff = (lane >= 16) ? 8 : 0;
#pragma unroll
  for (int i = 0; i < 4; ++i)
#pragma unroll
    for (int n = 0; n < 2; ++n) {
      const int nc = nBase + n * 16 + (lane & 15);
#pragma unroll
      for (int j = 0; j < 8; ++j) {
        const int m = mBase + i * 16 + j + mOff;
        xp[(size_t)m * NCOLS + nc] = acc[i][n][j];
      }
    }
}

// ---------------------------------------------------------------------------
// Kernel 3: persistent recurrent kernel. 16 blocks x 256 threads (8 waves).
// LDS: 128KB Wh slice (preloaded once, async) + 64KB hm stage (TDM per step).
// ---------------------------------------------------------------------------
__global__ __launch_bounds__(256) void lstm_rec_kernel(
    const float*  __restrict__ xp,     // [T][B][2048]
    const __bf16* __restrict__ Whb,    // [2048][512]
    const float*  __restrict__ bsum,   // [2048]
    const float*  __restrict__ zh,     // [512]
    __bf16*       __restrict__ hmbuf,  // 2 x [64][512] ping-pong
    unsigned*     __restrict__ syncc,
    float* __restrict__ out_o, float* __restrict__ out_h, float* __restrict__ out_c) {
  __shared__ __bf16 whLds[128 * HDIM];   // 128 KB: rows r = g*32 + kloc
  __shared__ __bf16 hmLds[BATCH * HDIM]; //  64 KB

  const int lane  = threadIdx.x & 31;
  const int wave  = threadIdx.x >> 5;
  const int rtile = wave >> 1;                  // 0..3  (batch rows 16*rtile..)
  const int kt    = wave & 1;                   // 0..1
  const int k0    = blockIdx.x * 32;            // this block's H slice
  const int kcol  = k0 + kt * 16 + (lane & 15); // lane's H column
  const int mOff  = (lane >= 16) ? 8 : 0;

  const unsigned whBase = (unsigned)(unsigned long long)(uintptr_t)&whLds[0];
  const unsigned hmBase = (unsigned)(unsigned long long)(uintptr_t)&hmLds[0];

  // ---- one-time: async-stage this block's Wh slice (128 rows x 512) ----
#pragma unroll
  for (int it = 0; it < 32; ++it) {
    const int chunk = it * 256 + (int)threadIdx.x;   // 8192 chunks of 16B
    const int r     = chunk >> 6;                    // local row 0..127
    const int co    = (chunk & 63) * 8;              // bf16 offset in row
    const int g     = r >> 5;
    const int kloc  = r & 31;
    unsigned long long ga =
        (unsigned long long)(Whb + ((size_t)(g * HDIM + k0 + kloc) * HDIM + co));
    unsigned la = whBase + (unsigned)chunk * 16u;
    asm volatile("global_load_async_to_lds_b128 %0, %1, off"
                 :: "v"(la), "v"(ga) : "memory");
  }
  asm volatile("s_wait_asynccnt 0" ::: "memory");
  __syncthreads();

  float bs0 = bsum[0 * HDIM + kcol];
  float bs1 = bsum[1 * HDIM + kcol];
  float bs2 = bsum[2 * HDIM + kcol];
  float bs3 = bsum[3 * HDIM + kcol];
  const float zhv = zh[kcol];

  v8f c_state = {0.f, 0.f, 0.f, 0.f, 0.f, 0.f, 0.f, 0.f};

  for (int t = 0; t < T_STEPS; ++t) {
    const __bf16* rd = hmbuf + (size_t)(t & 1) * (BATCH * HDIM);
    __bf16*       wr = hmbuf + (size_t)((t + 1) & 1) * (BATCH * HDIM);

    // ---- stage hm (64x512 bf16 = 64KB) into LDS via TDM ----
#if __has_builtin(__builtin_amdgcn_tensor_load_to_lds)
    if (wave == 0) {
      // D# group0: count=1 | lds_addr | 57b global addr | type=2
      const unsigned long long ga = (unsigned long long)rd;
      u32x4 g0 = { 1u, hmBase, (unsigned)ga,
                   (unsigned)(ga >> 32) | (2u << 30) };
      // D# group1: 2-D tensor of 8-byte elems: dim0=128/row, dim1=64 rows,
      // stride0=128; tile == tensor; wg_mask=0 (not in cluster).
      i32x8 g1 = { (int)(3u << 16),      // data_size=3 (8B)
                   (int)(128u << 16),    // tensor_dim0[15:0]=128
                   (int)(64u << 16),     // tensor_dim1[15:0]=64
                   (int)(128u << 16),    // tile_dim0=128
                   (int)64,              // tile_dim1=64
                   (int)128,             // tensor_dim0_stride=128
                   0, 0 };
      i32x4 gz4 = { 0, 0, 0, 0 };
      i32x8 gz8 = { 0, 0, 0, 0, 0, 0, 0, 0 };
      __builtin_amdgcn_tensor_load_to_lds(g0, g1, gz4, gz4, gz8, 0);
      __builtin_amdgcn_s_wait_tensorcnt(0);
    }
#else
#pragma unroll
    for (int it = 0; it < 16; ++it) {
      const int chunk = it * 256 + (int)threadIdx.x;
      unsigned long long ga = (unsigned long long)(rd + (size_t)chunk * 8);
      unsigned la = hmBase + (unsigned)chunk * 16u;
      asm volatile("global_load_async_to_lds_b128 %0, %1, off"
                   :: "v"(la), "v"(ga) : "memory");
    }
    asm volatile("s_wait_asynccnt 0" ::: "memory");
#endif
    __syncthreads();

    // prefetch this step's scattered xp gate rows (hide HBM latency)
    const size_t tb = (size_t)t * BATCH;
#pragma unroll
    for (int g = 0; g < 4; ++g) {
#pragma unroll
      for (int j = 0; j < 8; j += 4) {
        const int m = rtile * 16 + j + mOff;
        __builtin_prefetch(&xp[(tb + m) * (size_t)NCOLS + g * HDIM + kcol], 0, 3);
      }
    }

    // ---- 4 gate GEMMs over K=512: A from hmLds, B from whLds ----
    v8f acc[4];
#pragma unroll
    for (int g = 0; g < 4; ++g) acc[g] = (v8f){0.f,0.f,0.f,0.f,0.f,0.f,0.f,0.f};
#pragma unroll 4
    for (int kk = 0; kk < HDIM; kk += 32) {
      v16bf a = load_frag_rm(hmLds, rtile * 16, HDIM, kk, lane);
#pragma unroll
      for (int g = 0; g < 4; ++g) {
        v16bf b = load_frag_rm(whLds, g * 32 + kt * 16, HDIM, kk, lane);
        acc[g] = __builtin_amdgcn_wmma_f32_16x16x32_bf16(false, a, false, b,
                                                         (short)0, acc[g], false, false);
      }
    }

    // ---- gates + cell update + outputs (c stays in registers) ----
#pragma unroll
    for (int j = 0; j < 8; ++j) {
      const int    m    = rtile * 16 + j + mOff;      // batch row
      const size_t xrow = (tb + m) * (size_t)NCOLS;
      const float zi = acc[0][j] + xp[xrow + 0 * HDIM + kcol] + bs0;
      const float zf = acc[1][j] + xp[xrow + 1 * HDIM + kcol] + bs1;
      const float zo = acc[2][j] + xp[xrow + 2 * HDIM + kcol] + bs2;
      const float zg = acc[3][j] + xp[xrow + 3 * HDIM + kcol] + bs3;
      const float ig = fast_sigmoid(zi);
      const float fg = fast_sigmoid(zf);
      const float og = fast_sigmoid(zo);
      const float gg = fast_tanh(zg);
      const float cn = fg * c_state[j] + ig * gg;
      const float hn = og * fast_tanh(cn);
      c_state[j] = cn;
      const size_t oidx = (tb + m) * (size_t)HDIM + kcol;
      out_o[oidx] = og;
      out_h[oidx] = hn;
      out_c[oidx] = cn;
      wr[(size_t)m * HDIM + kcol] = (__bf16)(hn * zhv);
    }

    // ---- grid barrier (all hm writes visible before next step reads) ----
    __threadfence();
    __syncthreads();
    if (threadIdx.x == 0) {
      __hip_atomic_fetch_add(syncc, 1u, __ATOMIC_RELEASE, __HIP_MEMORY_SCOPE_AGENT);
      const unsigned target = (unsigned)(t + 1) * gridDim.x;
      while (__hip_atomic_load(syncc, __ATOMIC_ACQUIRE, __HIP_MEMORY_SCOPE_AGENT) < target) {
        __builtin_amdgcn_s_sleep(2);
      }
    }
    __syncthreads();
  }
}

// ---------------------------------------------------------------------------
// Launch
// ---------------------------------------------------------------------------
extern "C" void kernel_launch(void* const* d_in, const int* in_sizes, int n_in,
                              void* d_out, int out_size, void* d_ws, size_t ws_size,
                              hipStream_t stream) {
  (void)in_sizes; (void)n_in; (void)out_size; (void)ws_size;
  const float* x  = (const float*)d_in[0];
  // d_in[1] = h0 (zeros, implicit)
  const float* zx = (const float*)d_in[2];
  const float* zh = (const float*)d_in[3];
  const float* Wx = (const float*)d_in[4];
  const float* bx = (const float*)d_in[5];
  const float* Wh = (const float*)d_in[6];
  const float* bh = (const float*)d_in[7];

  float* out   = (float*)d_out;
  float* out_o = out;
  float* out_h = out + (size_t)T_STEPS * BATCH * HDIM;
  float* out_c = out_h + (size_t)T_STEPS * BATCH * HDIM;

  char* ws = (char*)d_ws;
  __bf16*  xb    = (__bf16*)ws;               ws += (size_t)T_STEPS * BATCH * IDIM * 2;
  __bf16*  Wxb   = (__bf16*)ws;               ws += (size_t)4 * HDIM * IDIM * 2;
  __bf16*  Whb   = (__bf16*)ws;               ws += (size_t)4 * HDIM * HDIM * 2;
  float*   bsum  = (float*)ws;                ws += (size_t)NCOLS * 4;
  __bf16*  hmbuf = (__bf16*)ws;               ws += (size_t)2 * BATCH * HDIM * 2;
  unsigned* sync = (unsigned*)ws;             ws += 256;
  float*   xp    = (float*)ws;                // 268 MB

  init_kernel<<<(BATCH * HDIM + 255) / 256, 256, 0, stream>>>(hmbuf, sync);

  convert_kernel<<<2048, 256, 0, stream>>>(x, zx, Wx, bx, Wh, bh, xb, Wxb, Whb, bsum);

  // 32768 units of 4x2 tiles, 8 units (waves) per block
  gemm_xp_kernel<<<4096, 256, 0, stream>>>(xb, Wxb, xp);

  lstm_rec_kernel<<<NBLK, 256, 0, stream>>>(xp, Whb, bsum, zh, hmbuf, sync,
                                            out_o, out_h, out_c);
}
